// MacroFlow_29978871726094
// MI455X (gfx1250) — matrix operand order repaired and checked by reading, not compiled, not run on hardware
//
#include <hip/hip_runtime.h>
#include <hip/hip_bf16.h>
#include <math.h>

typedef __attribute__((ext_vector_type(16))) __bf16 v16bf;
typedef __attribute__((ext_vector_type(4)))  __bf16 v4bf;
typedef __attribute__((ext_vector_type(8)))  float  v8f;

#define BB   2048
#define DD   8192
#define DMH  4096
#define HH   1024
#define IN1  5120
#define NFL  8

__device__ __forceinline__ __bf16 to_bf16(float f) { return (__bf16)f; }

// ---------------- small helper kernels ----------------

__global__ void k_zero_logdet(float* logdet) {
    int i = blockIdx.x * blockDim.x + threadIdx.x;
    if (i < BB) logdet[i] = 0.f;
}

__global__ void k_cond_fill(const int* __restrict__ scen,
                            const float* __restrict__ emb,
                            __bf16* __restrict__ act1) {
    int gid = blockIdx.x * blockDim.x + threadIdx.x;   // B*H threads
    int row = gid >> 10;
    int j   = gid & 1023;
    int s   = scen[row];
    act1[(size_t)row * IN1 + DMH + j] = to_bf16(emb[s * HH + j]);
}

__global__ void k_sum_als(const float* __restrict__ als, float* __restrict__ out) {
    __shared__ float red[256];
    float s = 0.f;
    for (int j = threadIdx.x; j < DD; j += 256) s += als[j];
    red[threadIdx.x] = s;
    __syncthreads();
    for (int off = 128; off > 0; off >>= 1) {
        if (threadIdx.x < off) red[threadIdx.x] += red[threadIdx.x + off];
        __syncthreads();
    }
    if (threadIdx.x == 0) out[0] = red[0];
}

// z' = (z + bias) * exp(ls); write z' (f32) and masked half -> bf16 act1
__global__ void k_prep(const float* __restrict__ zin, float* __restrict__ z,
                       const float* __restrict__ abias, const float* __restrict__ als,
                       __bf16* __restrict__ act1, int parity) {
    int t  = blockIdx.x * blockDim.x + threadIdx.x;
    int e0 = t * 4;
    int row = e0 >> 13;          // / 8192
    int col = e0 & (DD - 1);
    float4 zi = *(const float4*)(zin + (size_t)row * DD + col);
    float4 b4 = *(const float4*)(abias + col);
    float4 l4 = *(const float4*)(als + col);
    float4 zo;
    zo.x = (zi.x + b4.x) * __expf(l4.x);
    zo.y = (zi.y + b4.y) * __expf(l4.y);
    zo.z = (zi.z + b4.z) * __expf(l4.z);
    zo.w = (zi.w + b4.w) * __expf(l4.w);
    *(float4*)(z + (size_t)row * DD + col) = zo;
    // masked (unchanged) half feeds the conditioner MLP
    float m0 = parity ? zo.y : zo.x;
    float m1 = parity ? zo.w : zo.z;
    __bf16* ap = act1 + (size_t)row * IN1 + (col >> 1);
    ap[0] = to_bf16(m0);
    ap[1] = to_bf16(m1);
}

// y_u = x_u * exp(s) + t ; log_det[row] += sum(s) + sum_als
__global__ void k_couple(float* __restrict__ z,
                         const float* __restrict__ sbuf, const float* __restrict__ tbuf,
                         float* __restrict__ logdet, const float* __restrict__ sum_als,
                         int parity) {
    __shared__ float red[256];
    int row = blockIdx.x;
    float local = 0.f;
    for (int j = threadIdx.x; j < DMH; j += 256) {
        float sv = sbuf[(size_t)row * DMH + j];
        float tv = tbuf[(size_t)row * DMH + j];
        int uc = 2 * j + (parity ? 0 : 1);     // update the *other* half
        float xu = z[(size_t)row * DD + uc];
        z[(size_t)row * DD + uc] = xu * __expf(sv) + tv;
        local += sv;
    }
    red[threadIdx.x] = local;
    __syncthreads();
    for (int off = 128; off > 0; off >>= 1) {
        if (threadIdx.x < off) red[threadIdx.x] += red[threadIdx.x + off];
        __syncthreads();
    }
    if (threadIdx.x == 0) logdet[row] += red[0] + sum_als[0];
}

// ---------------- weight convert/transpose ----------------
// W f32 [K][N]  ->  Wb bf16 [N][K]   (32x32 LDS tile, padded to kill bank conflicts)
__global__ __launch_bounds__(256)
void k_wconv(const float* __restrict__ W, __bf16* __restrict__ Wb, int K, int N) {
    __shared__ float tile[32][33];
    const int k0 = blockIdx.x * 32;
    const int n0 = blockIdx.y * 32;
    const int t  = threadIdx.x;
    const int kl = t >> 5;          // 0..7
    const int nl = t & 31;
#pragma unroll
    for (int r = 0; r < 4; ++r)
        tile[kl + 8 * r][nl] = W[(size_t)(k0 + kl + 8 * r) * N + (n0 + nl)];
    __syncthreads();
    const int nw = t >> 3;          // 0..31
    const int kw = (t & 7) * 4;     // 0,4,...,28
    v4bf o;
#pragma unroll
    for (int j = 0; j < 4; ++j) o[j] = to_bf16(tile[kw + j][nw]);
    *(v4bf*)(Wb + (size_t)(n0 + nw) * K + (k0 + kw)) = o;
}

// ---------------- WMMA GEMM ----------------
// C[M=2048, N] = A[M, K](bf16, lda) * Wb[N, K](bf16) + bias, epilogue:
//   MODE 0: leaky_relu(0.2) -> bf16 out
//   MODE 1: tanh(.)*sf      -> f32 out
//   MODE 2: identity        -> f32 out
// Block tile 128x128, BK=32, 256 threads = 8 waves, wave tile 32x64 (2x4 WMMA frags).
// LDS tiles in fragment-native layout (frag f -> 32 lanes x 16 bf16, 32B/lane).
// Global loads for K-slice k+1 are issued between the LDS stores and the compute
// barrier of slice k, overlapping VMEM with ds_load + v_wmma.
template <int MODE>
__global__ __launch_bounds__(256)
void k_gemm(const __bf16* __restrict__ A, int lda,
            const __bf16* __restrict__ Wb, int K, int N,
            const float* __restrict__ bias,
            void* __restrict__ outv,
            const float* __restrict__ sfp) {
    __shared__ alignas(32) __bf16 lA[8 * 512];   // 8 KB
    __shared__ alignas(32) __bf16 lB[8 * 512];   // 8 KB

    const int tid  = threadIdx.x;
    const int lane = tid & 31;
    const int wave = tid >> 5;
    const int mw = wave & 3;          // 4 wave rows  (M)
    const int nw = wave >> 2;         // 2 wave cols  (N)
    const int blockM = blockIdx.x * 128;
    const int blockN = blockIdx.y * 128;

    // fill-side mapping: one (frag, lane) per thread
    const int f  = tid >> 5;              // frag 0..7
    const int fl = tid & 31;
    // A frag (16x32 bf16): lane<16 -> K 0-7 & 16-23 ; lane>=16 -> K 8-15 & 24-31
    const int aRow  = blockM + f * 16 + (fl & 15);
    const int aKoff = (fl >> 4) << 3;     // 0 or 8
    // B frag (32x16 bf16): lane<16 -> K 0-7 & 16-23 ; lane>=16 -> K 8-15 & 24-31
    const int bCol  = blockN + f * 16 + (fl & 15);
    const int bKoff = (fl >> 4) << 3;     // 0 or 8

    const __bf16* aPtr = A  + (size_t)aRow * lda + aKoff;
    const __bf16* bPtr = Wb + (size_t)bCol * K   + bKoff;
    uint4* dA = (uint4*)(lA + f * 512 + fl * 16);
    uint4* dB = (uint4*)(lB + f * 512 + fl * 16);

    v8f acc[2][4];
    v8f zero = {};
#pragma unroll
    for (int i = 0; i < 2; ++i)
#pragma unroll
        for (int j = 0; j < 4; ++j) acc[i][j] = zero;

    // preload K-slice 0
    uint4 a0 = *(const uint4*)(aPtr);
    uint4 a1 = *(const uint4*)(aPtr + 16);
    uint4 b0 = *(const uint4*)(bPtr);
    uint4 b1 = *(const uint4*)(bPtr + 16);

    for (int k0 = 0; k0 < K; k0 += 32) {
        __syncthreads();               // previous slice fully consumed
        dA[0] = a0; dA[1] = a1;
        dB[0] = b0; dB[1] = b1;
        if (k0 + 32 < K) {             // uniform branch: preload next slice
            a0 = *(const uint4*)(aPtr + k0 + 32);
            a1 = *(const uint4*)(aPtr + k0 + 48);
            b0 = *(const uint4*)(bPtr + k0 + 32);
            b1 = *(const uint4*)(bPtr + k0 + 48);
        }
        if (k0 + 64 < K) {             // warm L2/L0 two slices ahead
            __builtin_prefetch(aPtr + k0 + 64, 0, 0);
            __builtin_prefetch(bPtr + k0 + 64, 0, 0);
        }
        __syncthreads();               // LDS tile ready

        v16bf af[2], bfr[4];
#pragma unroll
        for (int i = 0; i < 2; ++i)
            af[i] = *(const v16bf*)(lA + (mw * 2 + i) * 512 + lane * 16);
#pragma unroll
        for (int j = 0; j < 4; ++j)
            bfr[j] = *(const v16bf*)(lB + (nw * 4 + j) * 512 + lane * 16);

#pragma unroll
        for (int i = 0; i < 2; ++i)
#pragma unroll
            for (int j = 0; j < 4; ++j)
                acc[i][j] = __builtin_amdgcn_wmma_f32_16x16x32_bf16(
                    false, af[i], false, bfr[j], (short)0, acc[i][j], false, false);
    }

    // epilogue: C/D layout = lane<16: N=lane, M=r ; lane>=16: N=lane-16, M=8+r
    const int nBase = blockN + nw * 64;
    const int mBase = blockM + mw * 32;
    const int nLane = lane & 15;
    const int mOff  = (lane >> 4) << 3;
    float sfv = (MODE == 1) ? sfp[0] : 0.f;
#pragma unroll
    for (int i = 0; i < 2; ++i) {
#pragma unroll
        for (int j = 0; j < 4; ++j) {
            int gn = nBase + j * 16 + nLane;
            float bv = bias[gn];
#pragma unroll
            for (int r = 0; r < 8; ++r) {
                int gm = mBase + i * 16 + mOff + r;
                float v = acc[i][j][r] + bv;
                if (MODE == 0) {
                    v = v > 0.f ? v : 0.2f * v;              // leaky_relu(0.2)
                    ((__bf16*)outv)[(size_t)gm * N + gn] = to_bf16(v);
                } else if (MODE == 1) {
                    ((float*)outv)[(size_t)gm * N + gn] = tanhf(v) * sfv;
                } else {
                    ((float*)outv)[(size_t)gm * N + gn] = v;
                }
            }
        }
    }
}

// ---------------- driver ----------------

extern "C" void kernel_launch(void* const* d_in, const int* in_sizes, int n_in,
                              void* d_out, int out_size, void* d_ws, size_t ws_size,
                              hipStream_t stream) {
    const float* x    = (const float*)d_in[0];
    const int*   scen = (const int*)d_in[1];
    const float* emb  = (const float*)d_in[2];
    const float* als  = (const float*)d_in[3];
    const float* ab   = (const float*)d_in[4];
    const float* W1   = (const float*)d_in[5];
    const float* b1   = (const float*)d_in[6];
    const float* W2   = (const float*)d_in[7];
    const float* b2   = (const float*)d_in[8];
    const float* Ws   = (const float*)d_in[9];
    const float* bs   = (const float*)d_in[10];
    const float* Wt   = (const float*)d_in[11];
    const float* bt   = (const float*)d_in[12];
    const float* sf   = (const float*)d_in[13];

    float* z      = (float*)d_out;                       // [B, D]
    float* logdet = z + (size_t)BB * DD;                 // [B]

    char* ws = (char*)d_ws;
    __bf16* act1    = (__bf16*)(ws);                     // [B, 5120] bf16 (x_c | cond)
    __bf16* h1      = (__bf16*)(ws + 20971520);          // [B, 1024] bf16
    __bf16* h2      = (__bf16*)(ws + 25165824);          // [B, 1024] bf16
    float*  sbuf    = (float*)(ws + 29360128);           // [B, 4096] f32
    float*  tbuf    = (float*)(ws + 62914560);           // [B, 4096] f32
    float*  sum_als = (float*)(ws + 96468992);           // scalar
    __bf16* wbase   = (__bf16*)(ws + 100663296);         // per-flow bf16 weights [N][K]
    __bf16* wb1 = wbase;                                 // [1024][5120]
    __bf16* wb2 = wbase + 5242880;                       // [1024][1024]
    __bf16* wbs = wbase + 6291456;                       // [4096][1024]
    __bf16* wbt = wbase + 10485760;                      // [4096][1024]

    k_zero_logdet<<<8, 256, 0, stream>>>(logdet);
    k_cond_fill<<<(BB * HH) / 256, 256, 0, stream>>>(scen, emb, act1);

    for (int i = 0; i < NFL; ++i) {
        int parity = i & 1;
        const float* zin = (i == 0) ? x : z;

        k_sum_als<<<1, 256, 0, stream>>>(als + (size_t)i * DD, sum_als);
        k_prep<<<(BB * DD) / 1024, 256, 0, stream>>>(zin, z, ab + (size_t)i * DD,
                                                     als + (size_t)i * DD, act1, parity);

        // convert this flow's weights: f32 [K][N] -> bf16 [N][K]
        k_wconv<<<dim3(IN1 / 32, HH / 32), 256, 0, stream>>>(
            W1 + (size_t)i * IN1 * HH, wb1, IN1, HH);
        k_wconv<<<dim3(HH / 32, HH / 32), 256, 0, stream>>>(
            W2 + (size_t)i * HH * HH, wb2, HH, HH);
        k_wconv<<<dim3(HH / 32, DMH / 32), 256, 0, stream>>>(
            Ws + (size_t)i * HH * DMH, wbs, HH, DMH);
        k_wconv<<<dim3(HH / 32, DMH / 32), 256, 0, stream>>>(
            Wt + (size_t)i * HH * DMH, wbt, HH, DMH);

        // h1 = leaky_relu([x_c, cond] @ W1 + b1) : 2048x1024, K=5120
        k_gemm<0><<<dim3(16, 8), 256, 0, stream>>>(
            act1, IN1, wb1, IN1, HH, b1 + (size_t)i * HH, (void*)h1, sf + i);

        // h2 = leaky_relu(h1 @ W2 + b2) : 2048x1024, K=1024
        k_gemm<0><<<dim3(16, 8), 256, 0, stream>>>(
            h1, HH, wb2, HH, HH, b2 + (size_t)i * HH, (void*)h2, sf + i);

        // s = tanh(h2 @ Ws + bs) * sf : 2048x4096, K=1024
        k_gemm<1><<<dim3(16, 32), 256, 0, stream>>>(
            h2, HH, wbs, HH, DMH, bs + (size_t)i * DMH, (void*)sbuf, sf + i);

        // t = h2 @ Wt + bt : 2048x4096, K=1024
        k_gemm<2><<<dim3(16, 32), 256, 0, stream>>>(
            h2, HH, wbt, HH, DMH, bt + (size_t)i * DMH, (void*)tbuf, sf + i);

        k_couple<<<BB, 256, 0, stream>>>(z, sbuf, tbuf, logdet, sum_als, parity);
    }
    (void)in_sizes; (void)n_in; (void)out_size; (void)ws_size;
}
